// LSTMModel_69295002353800
// MI455X (gfx1250) — matrix operand order repaired and checked by reading, not compiled, not run on hardware
//
#include <hip/hip_runtime.h>
#include <hip/hip_fp16.h>

typedef __attribute__((ext_vector_type(16))) _Float16 v16h;
typedef __attribute__((ext_vector_type(8)))  _Float16 v8h;
typedef __attribute__((ext_vector_type(8)))  float    v8f;
typedef __attribute__((ext_vector_type(4)))  unsigned int v4u;
typedef __attribute__((ext_vector_type(4)))  int      v4i;
typedef __attribute__((ext_vector_type(8)))  int      v8i;

#define NWG 32   // workgroups in persistent recurrent kernel (each owns 16 cols of H)

// ---------------------------------------------------------------------------
// Elementwise helpers
// ---------------------------------------------------------------------------
__global__ void f32_to_f16_k(const float* __restrict__ in, _Float16* __restrict__ out, int n) {
    int i = blockIdx.x * blockDim.x + threadIdx.x;
    if (i < n) out[i] = (_Float16)in[i];
}

__global__ void vec_add_f32(const float* __restrict__ a, const float* __restrict__ b,
                            float* __restrict__ o, int n) {
    int i = blockIdx.x * blockDim.x + threadIdx.x;
    if (i < n) o[i] = a[i] + b[i];
}

// ---------------------------------------------------------------------------
// Pack W[N][K] (f32, row-major) into WMMA B-fragment order (f16):
//   fragment for (tileN tn, kChunk kc) = 512 halfs; lane L, element e holds
//   B[k][n] = W[n][k] with n = tn*16 + (L%16), k = kc*32 + 8*(L/16) + e (e<8), kc*32 + 16 + 8*(L/16) + (e-8) (e>=8).
// One wave per (tn,kc) fragment; lane reads two runs of 8 floats, 16 apart, 1x32B store.
// ---------------------------------------------------------------------------
__global__ void pack_b(const float* __restrict__ W, _Float16* __restrict__ out, int N, int K) {
    int gw   = (blockIdx.x * blockDim.x + threadIdx.x) >> 5;
    int lane = threadIdx.x & 31;
    int kch  = K >> 5;
    int total = (N >> 4) * kch;
    if (gw >= total) return;
    int tn = gw / kch;
    int kc = gw - tn * kch;
    int n  = tn * 16 + (lane & 15);
    int k0 = kc * 32 + 8 * (lane >> 4);
    const float* src = W + (size_t)n * K + k0;
    v16h tmp;
#pragma unroll
    for (int e = 0; e < 16; ++e) tmp[e] = (_Float16)src[e < 8 ? e : e + 8];
    *(v16h*)(out + (size_t)gw * 512 + lane * 16) = tmp;
}

// ---------------------------------------------------------------------------
// A-fragment load: 16x32 f16 tile per ISA layout (two 16B loads per lane).
// ---------------------------------------------------------------------------
__device__ __forceinline__ v16h load_a_frag(const _Float16* __restrict__ rowp, int koff) {
    v8h a0 = *(const v8h*)(rowp + koff);
    v8h a1 = *(const v8h*)(rowp + koff + 16);
    v16h a;
#pragma unroll
    for (int i = 0; i < 8; ++i) { a[i] = a0[i]; a[i + 8] = a1[i]; }
    return a;
}

// ---------------------------------------------------------------------------
// Generic WMMA GEMM:  C[M,N] = op(A[M,K]) * Bpacked + bias
//   One wave computes a 16x64 strip: 4 N-tiles sharing one A fragment per
//   k-chunk -> 4 independent v_wmma per iteration.  RELU_A / C_F16 are
//   compile-time so the non-ReLU GEMMs carry zero extra VALU.
// ---------------------------------------------------------------------------
template<bool RELU_A, bool C_F16>
__global__ void gemm16_t(const _Float16* __restrict__ A, const _Float16* __restrict__ Bpk,
                         const float* __restrict__ bias, void* __restrict__ Cout,
                         int M, int N, int K) {
    int gw   = (blockIdx.x * blockDim.x + threadIdx.x) >> 5;
    int lane = threadIdx.x & 31;
    int ntg  = N >> 6;                 // groups of 4 N-tiles
    int tm   = gw / ntg;
    int tg   = gw - tm * ntg;
    if (tm >= (M >> 4)) return;
    int tn   = tg * 4;
    int kch  = K >> 5;
    int nl   = lane & 15;
    const _Float16* arow = A + (size_t)(tm * 16 + nl) * K + 8 * (lane >> 4);
    const _Float16* bp0 = Bpk + (size_t)(tn + 0) * kch * 512 + lane * 16;
    const _Float16* bp1 = Bpk + (size_t)(tn + 1) * kch * 512 + lane * 16;
    const _Float16* bp2 = Bpk + (size_t)(tn + 2) * kch * 512 + lane * 16;
    const _Float16* bp3 = Bpk + (size_t)(tn + 3) * kch * 512 + lane * 16;
    v8f acc0 = {}, acc1 = {}, acc2 = {}, acc3 = {};
    const v16h zero16 = {};
    for (int kc = 0; kc < kch; ++kc) {
        v16h a = load_a_frag(arow, kc * 32);
        if (RELU_A) a = __builtin_elementwise_max(a, zero16);
        v16h b0 = *(const v16h*)(bp0 + (size_t)kc * 512);
        v16h b1 = *(const v16h*)(bp1 + (size_t)kc * 512);
        v16h b2 = *(const v16h*)(bp2 + (size_t)kc * 512);
        v16h b3 = *(const v16h*)(bp3 + (size_t)kc * 512);
        acc0 = __builtin_amdgcn_wmma_f32_16x16x32_f16(false, a, false, b0, (short)0, acc0, false, false);
        acc1 = __builtin_amdgcn_wmma_f32_16x16x32_f16(false, a, false, b1, (short)0, acc1, false, false);
        acc2 = __builtin_amdgcn_wmma_f32_16x16x32_f16(false, a, false, b2, (short)0, acc2, false, false);
        acc3 = __builtin_amdgcn_wmma_f32_16x16x32_f16(false, a, false, b3, (short)0, acc3, false, false);
    }
    int mbase = tm * 16 + 8 * (lane >> 4);
#pragma unroll
    for (int i = 0; i < 4; ++i) {
        v8f ac = (i == 0) ? acc0 : (i == 1) ? acc1 : (i == 2) ? acc2 : acc3;
        float bv = bias[(tn + i) * 16 + nl];
        if (C_F16) {
            _Float16* C = (_Float16*)Cout;
#pragma unroll
            for (int r = 0; r < 8; ++r)
                C[(size_t)(mbase + r) * N + (tn + i) * 16 + nl] = (_Float16)(ac[r] + bv);
        } else {
            float* C = (float*)Cout;
#pragma unroll
            for (int r = 0; r < 8; ++r)
                C[(size_t)(mbase + r) * N + (tn + i) * 16 + nl] = ac[r] + bv;
        }
    }
}

// ---------------------------------------------------------------------------
// Persistent LSTM recurrence with TDM-staged h.
//   gx   : [B=64, T, 4H=2048] f16,  Wpk: Whh packed,  h_out: [B,T,H] f16
// NWG=32 WGs x 256 threads.  LDS (144KB dynamic of 320KB/WGP):
//   [0,16K)    gate exchange (4x64x16 f32)
//   [16K,80K)  Whh slice, staged once, reused 512x
//   [80K,144K) h_{t-1} tile (64x512 f16), fetched per step by the Tensor Data
//              Mover: one tensor_load_to_lds describing the strided 2-D tile
//              (tile 512x64, row stride T*512, 2-byte elements), waited with
//              s_wait_tensorcnt.  WMMA A/B fragments then come from LDS.
// ---------------------------------------------------------------------------
__device__ __forceinline__ float sigm_f(float x)    { return 1.0f / (1.0f + __expf(-x)); }
__device__ __forceinline__ float tanh_fast(float x) { return 2.0f / (1.0f + __expf(-2.0f * x)) - 1.0f; }

__global__ void lstm_rec(const _Float16* __restrict__ gx, const _Float16* __restrict__ Wpk,
                         _Float16* __restrict__ h_out, int T,
                         int* __restrict__ bar_cnt, int* __restrict__ bar_gen) {
    extern __shared__ char smem_raw[];
    float*    g_lds = (float*)smem_raw;                       // 16KB
    _Float16* b_lds = (_Float16*)(smem_raw + 16 * 1024);      // 64KB
    _Float16* h_lds = (_Float16*)(smem_raw + 80 * 1024);      // 64KB

    const int w    = blockIdx.x;
    const int n0   = w * 16;
    const int tid  = threadIdx.x;
    const int lane = tid & 31;
    const int wave = tid >> 5;
    const int mt   = wave & 3;
    const int ga   = (wave >> 2) * 2;
    const int nl   = lane & 15;
    const int kh   = 8 * (lane >> 4);

    // --- one-time stage of this WG's Whh slice into LDS (reused 512x) ---
    for (int g = 0; g < 4; ++g) {
        const v16h* src = (const v16h*)(Wpk + (size_t)(g * 32 + w) * 16 * 512);
        v16h*       dst = (v16h*)(b_lds + (size_t)g * 8192);
        for (int i = tid; i < 512; i += 256) dst[i] = src[i];
    }
    __syncthreads();

    const _Float16* bl0 = b_lds + (size_t)(ga    ) * 8192 + lane * 16;
    const _Float16* bl1 = b_lds + (size_t)(ga + 1) * 8192 + lane * 16;
    const unsigned int h_lds_off = (unsigned int)(size_t)h_lds;   // LDS byte offset
    float creg[4] = {0.f, 0.f, 0.f, 0.f};
    int mygen = 0;

    for (int t = 0; t < T; ++t) {
        v8f acc0 = {}; v8f acc1 = {};
        if (t > 0) {
            // --- TDM: DMA h_{t-1} (64 rows x 512 halfs, row stride T*512) to LDS ---
            if (wave == 0) {
                unsigned long long gaddr =
                    (unsigned long long)(size_t)h_out + (unsigned long long)(t - 1) * 1024ull;
                v4u g0;
                g0[0] = 1u;                                    // count=1, user mode
                g0[1] = h_lds_off;                             // lds_addr (bytes)
                g0[2] = (unsigned int)gaddr;                   // global_addr[31:0]
                g0[3] = (unsigned int)(gaddr >> 32) | (2u << 30);  // addr[56:32] | type=2
                v8i g1;
                g1[0] = (int)(1u << 16);        // data_size=1 (2 bytes)
                g1[1] = (int)(512u << 16);      // tensor_dim0[15:0]=512
                g1[2] = (int)(64u << 16);       // dim0[31:16]=0 | tensor_dim1[15:0]=64
                g1[3] = (int)(512u << 16);      // dim1[31:16]=0 | tile_dim0=512
                g1[4] = (int)64;                // tile_dim1=64 | tile_dim2=0
                g1[5] = (int)(512 * 512);       // tensor_dim0_stride = T*512 elements
                g1[6] = 0;                      // stride[47:32] | dim1_stride lo
                g1[7] = 0;
                v4i z4 = {0, 0, 0, 0};
                v8i z8 = {0, 0, 0, 0, 0, 0, 0, 0};
                __builtin_amdgcn_tensor_load_to_lds(g0, g1, z4, z4, z8, 0);
                __builtin_amdgcn_s_wait_tensorcnt(0);
            }
            __syncthreads();
            const _Float16* arow = h_lds + (size_t)(mt * 16 + nl) * 512 + kh;
#pragma unroll 2
            for (int kc = 0; kc < 16; ++kc) {
                v16h a  = load_a_frag(arow, kc * 32);
                v16h b0 = *(const v16h*)(bl0 + (size_t)kc * 512);
                v16h b1 = *(const v16h*)(bl1 + (size_t)kc * 512);
                acc0 = __builtin_amdgcn_wmma_f32_16x16x32_f16(false, a, false, b0, (short)0, acc0, false, false);
                acc1 = __builtin_amdgcn_wmma_f32_16x16x32_f16(false, a, false, b1, (short)0, acc1, false, false);
            }
        }
        // add precomputed input contribution and stage gates in LDS
#pragma unroll
        for (int r = 0; r < 8; ++r) {
            int ml = mt * 16 + r + 8 * (lane >> 4);           // batch row
            size_t base = ((size_t)ml * T + t) * 2048;
            float v0 = acc0[r] + (float)gx[base + (size_t)ga * 512 + n0 + nl];
            float v1 = acc1[r] + (float)gx[base + (size_t)(ga + 1) * 512 + n0 + nl];
            g_lds[ga * 1024 + ml * 16 + nl]       = v0;
            g_lds[(ga + 1) * 1024 + ml * 16 + nl] = v1;
            if (t + 1 < T) {  // pull next step's gx slice toward the caches early
                __builtin_prefetch(gx + base + 2048 + (size_t)ga * 512 + n0 + nl, 0, 1);
            }
        }
        __syncthreads();
        // elementwise cell update: each thread owns 4 (b, col) elements
#pragma unroll
        for (int j = 0; j < 4; ++j) {
            int idx = tid * 4 + j;                 // 0..1023 over 64x16
            int m   = idx >> 4;
            int nn  = idx & 15;
            float gi = sigm_f(g_lds[idx]);
            float gf = sigm_f(g_lds[1024 + idx]);
            float gg = tanh_fast(g_lds[2048 + idx]);
            float go = sigm_f(g_lds[3072 + idx]);
            float c  = gf * creg[j] + gi * gg;
            creg[j]  = c;
            float h  = go * tanh_fast(c);
            h_out[((size_t)m * T + t) * 512 + n0 + nn] = (_Float16)h;
        }
        // grid-wide barrier: release h_t, wait for all WGs
        __threadfence();
        __syncthreads();
        if (tid == 0) {
            ++mygen;
            int v = __hip_atomic_fetch_add(bar_cnt, 1, __ATOMIC_ACQ_REL, __HIP_MEMORY_SCOPE_AGENT);
            if (v == NWG - 1) {
                __hip_atomic_store(bar_cnt, 0, __ATOMIC_RELAXED, __HIP_MEMORY_SCOPE_AGENT);
                __hip_atomic_store(bar_gen, mygen, __ATOMIC_RELEASE, __HIP_MEMORY_SCOPE_AGENT);
            } else {
                while (__hip_atomic_load(bar_gen, __ATOMIC_ACQUIRE, __HIP_MEMORY_SCOPE_AGENT) < mygen) {
                    __builtin_amdgcn_s_sleep(1);
                }
            }
        }
        __syncthreads();
        __threadfence();
    }
}

// ---------------------------------------------------------------------------
// Host orchestration (graph-capturable: async memset + kernel launches only)
// ---------------------------------------------------------------------------
extern "C" void kernel_launch(void* const* d_in, const int* in_sizes, int n_in,
                              void* d_out, int out_size, void* d_ws, size_t ws_size,
                              hipStream_t stream) {
    (void)in_sizes; (void)n_in; (void)out_size; (void)ws_size;
    const float* x    = (const float*)d_in[0];
    const float* Wih0 = (const float*)d_in[1];
    const float* Whh0 = (const float*)d_in[2];
    const float* bih0 = (const float*)d_in[3];
    const float* bhh0 = (const float*)d_in[4];
    const float* Wih1 = (const float*)d_in[5];
    const float* Whh1 = (const float*)d_in[6];
    const float* bih1 = (const float*)d_in[7];
    const float* bhh1 = (const float*)d_in[8];
    const float* Wfc  = (const float*)d_in[9];
    const float* bfc  = (const float*)d_in[10];
    float* out = (float*)d_out;

    const int B = 64, T = 512, D = 128, H = 512;
    const int M = B * T;                 // 32768 GEMM rows
    char* p = (char*)d_ws;
    auto alloc = [&](size_t bytes) { char* r = p; p += (bytes + 255) & ~(size_t)255; return r; };
    _Float16* x16   = (_Float16*)alloc((size_t)M * D * 2);
    _Float16* h0    = (_Float16*)alloc((size_t)M * H * 2);
    _Float16* h1    = (_Float16*)alloc((size_t)M * H * 2);
    _Float16* gx0   = (_Float16*)alloc((size_t)M * 4 * H * 2);
    _Float16* gx1   = (_Float16*)alloc((size_t)M * 4 * H * 2);
    _Float16* Wih0p = (_Float16*)alloc((size_t)4 * H * D * 2);
    _Float16* Whh0p = (_Float16*)alloc((size_t)4 * H * H * 2);
    _Float16* Wih1p = (_Float16*)alloc((size_t)4 * H * H * 2);
    _Float16* Whh1p = (_Float16*)alloc((size_t)4 * H * H * 2);
    _Float16* Wfcp  = (_Float16*)alloc((size_t)128 * H * 2);
    float* b0s = (float*)alloc(4 * H * 4);
    float* b1s = (float*)alloc(4 * H * 4);
    int*   bar = (int*)alloc(512);        // bar[0]=count, bar[64]=generation

    // --- convert / pack / bias prep ---
    { int n = M * D; f32_to_f16_k<<<(n + 255) / 256, 256, 0, stream>>>(x, x16, n); }
    { int tw = (4 * H / 16) * (D / 32); pack_b<<<(tw * 32 + 255) / 256, 256, 0, stream>>>(Wih0, Wih0p, 4 * H, D); }
    { int tw = (4 * H / 16) * (H / 32); pack_b<<<(tw * 32 + 255) / 256, 256, 0, stream>>>(Whh0, Whh0p, 4 * H, H); }
    { int tw = (4 * H / 16) * (H / 32); pack_b<<<(tw * 32 + 255) / 256, 256, 0, stream>>>(Wih1, Wih1p, 4 * H, H); }
    { int tw = (4 * H / 16) * (H / 32); pack_b<<<(tw * 32 + 255) / 256, 256, 0, stream>>>(Whh1, Whh1p, 4 * H, H); }
    { int tw = (128 / 16) * (H / 32);   pack_b<<<(tw * 32 + 255) / 256, 256, 0, stream>>>(Wfc,  Wfcp, 128, H); }
    vec_add_f32<<<(4 * H + 255) / 256, 256, 0, stream>>>(bih0, bhh0, b0s, 4 * H);
    vec_add_f32<<<(4 * H + 255) / 256, 256, 0, stream>>>(bih1, bhh1, b1s, 4 * H);

    const int REC_SMEM = 144 * 1024;   // gates + Whh slice + TDM h tile

    // --- layer 0: gx0 = x·Wih0ᵀ + b, then recurrence ---
    { int waves = (M / 16) * (4 * H / 64);   // 4 N-tiles per wave
      gemm16_t<false, true><<<waves / 8, 256, 0, stream>>>(x16, Wih0p, b0s, gx0, M, 4 * H, D); }
    (void)hipMemsetAsync(bar, 0, 512, stream);
    lstm_rec<<<NWG, 256, REC_SMEM, stream>>>(gx0, Whh0p, h0, T, bar, bar + 64);

    // --- layer 1: gx1 = h0·Wih1ᵀ + b, then recurrence ---
    { int waves = (M / 16) * (4 * H / 64);
      gemm16_t<false, true><<<waves / 8, 256, 0, stream>>>(h0, Wih1p, b1s, gx1, M, 4 * H, H); }
    (void)hipMemsetAsync(bar, 0, 512, stream);
    lstm_rec<<<NWG, 256, REC_SMEM, stream>>>(gx1, Whh1p, h1, T, bar, bar + 64);

    // --- FC: out = relu(h1)·Wfcᵀ + bfc (ReLU folded into A-load) ---
    { int waves = (M / 16) * (128 / 64);
      gemm16_t<true, false><<<waves / 8, 256, 0, stream>>>(h1, Wfcp, bfc, out, M, 128, H); }
}
